// VectorQuantizer_55018531062723
// MI455X (gfx1250) — compile-verified, matrix operand order checked by reading, not run on hardware
//
#include <hip/hip_runtime.h>
#include <hip/hip_bf16.h>
#include <math.h>
#include <stdint.h>

#define N_ROWS 32768
#define DIM 512
#define K_CODES 4096
#define ROWS_PER_BLOCK 64
#define ARGMIN_THREADS 128   // 4 waves of 32

// B staging: 64 codes x 64 dims per chunk, hi+lo, double buffered
#define B_CODES 64
#define B_DIMS 64
#define N_KB (K_CODES / B_CODES)      // 64
#define N_DB (DIM / B_DIMS)           // 8
#define N_CHUNKS (N_KB * N_DB)        // 512

typedef __attribute__((ext_vector_type(16))) __bf16 v16bf;
typedef __attribute__((ext_vector_type(8)))  float  v8f;

union V16 { v16bf v; uint4 q[2]; };

// ---- workspace layout (bytes) ----
static constexpr size_t OFF_XH  = 0;
static constexpr size_t OFF_XL  = OFF_XH  + (size_t)N_ROWS * DIM * 2;
static constexpr size_t OFF_EH  = OFF_XL  + (size_t)N_ROWS * DIM * 2;
static constexpr size_t OFF_EL  = OFF_EH  + (size_t)K_CODES * DIM * 2;
static constexpr size_t OFF_EN2 = OFF_EL  + (size_t)K_CODES * DIM * 2;
static constexpr size_t OFF_IDX = OFF_EN2 + (size_t)K_CODES * 4;
static constexpr size_t OFF_CNT = OFF_IDX + (size_t)N_ROWS * 4;
static constexpr size_t OFF_ACC = OFF_CNT + (size_t)K_CODES * 4;

__device__ inline unsigned short f32_to_bf16_rn(float f) {
    unsigned int u = __float_as_uint(f);
    unsigned int r = (u + 0x7FFFu + ((u >> 16) & 1u)) >> 16;   // round-to-nearest-even
    return (unsigned short)r;
}
__device__ inline float bf16_to_f32(unsigned short h) {
    return __uint_as_float((unsigned int)h << 16);
}

// CDNA5 async global->LDS (ASYNCcnt-tracked, no VGPR round trip).
// lds_off: byte offset within wave's LDS; g: 64-bit global address.
__device__ inline void async_ld_b128(unsigned lds_off, const unsigned short* g) {
    asm volatile("global_load_async_to_lds_b128 %0, %1, off"
                 :: "v"(lds_off), "v"(g) : "memory");
}
__device__ inline void wait_asynccnt0() {
    asm volatile("s_wait_asynccnt 0x0" ::: "memory");
}

// ---------------- Kernel 1a: split fp32 -> bf16 hi + bf16 lo ----------------
__global__ void vq_split_bf16(const float* __restrict__ src,
                              unsigned short* __restrict__ hi,
                              unsigned short* __restrict__ lo, int n) {
    int i = blockIdx.x * blockDim.x + threadIdx.x;
    if (i < n) {
        float x = src[i];
        unsigned short h = f32_to_bf16_rn(x);
        float residual = x - bf16_to_f32(h);
        hi[i] = h;
        lo[i] = f32_to_bf16_rn(residual);
    }
}

// ---------------- Kernel 1b: per-code squared norms ----------------
__global__ void vq_enorm(const float* __restrict__ cb, float* __restrict__ en2) {
    int c = blockIdx.x * blockDim.x + threadIdx.x;
    if (c < K_CODES) {
        const float4* p = (const float4*)(cb + (size_t)c * DIM);
        float s = 0.f;
        #pragma unroll 4
        for (int d = 0; d < DIM / 4; ++d) {
            float4 v = p[d];
            s += v.x * v.x + v.y * v.y + v.z * v.z + v.w * v.w;
        }
        en2[c] = s;
    }
}

// ---------------- Kernel 1c: zero accumulators (every launch -> deterministic) ----
__global__ void vq_zero(int* __restrict__ counts, float* __restrict__ loss_acc) {
    int i = blockIdx.x * blockDim.x + threadIdx.x;
    if (i < K_CODES) counts[i] = 0;
    if (i == 0) *loss_acc = 0.f;
}

// ---------------- Kernel 2: WMMA split-bf16 distance GEMM + argmin ----------------
// dot(x,e) ~= xh*eh + xh*el + xl*eh  (3 bf16 WMMAs per 32-K step per 16x16 tile)
// A (64 rows) resident in LDS; B (codebook) streamed via async DMA into a
// double-buffered LDS chunk shared by all 4 waves.
__launch_bounds__(ARGMIN_THREADS)
__global__ void vq_argmin_wmma(const unsigned short* __restrict__ xh,
                               const unsigned short* __restrict__ xl,
                               const unsigned short* __restrict__ eh,
                               const unsigned short* __restrict__ el,
                               const float* __restrict__ en2,
                               int* __restrict__ idx_out) {
    __shared__ __align__(16) unsigned short sxh[ROWS_PER_BLOCK * DIM];   // 64 KB
    __shared__ __align__(16) unsigned short sxl[ROWS_PER_BLOCK * DIM];   // 64 KB
    __shared__ __align__(16) unsigned short sbh[2][B_CODES * B_DIMS];    // 2 x 8 KB
    __shared__ __align__(16) unsigned short sbl[2][B_CODES * B_DIMS];    // 2 x 8 KB

    const int tid  = threadIdx.x;
    const int lane = tid & 31;
    const int wave = tid >> 5;
    const int rowBase = blockIdx.x * ROWS_PER_BLOCK;

    // LDS byte offsets for the async DMA destination (flat addr low 32 bits
    // of an LDS pointer are the LDS offset).
    const unsigned ldsbh = (unsigned)(uintptr_t)(void*)&sbh[0][0];
    const unsigned ldsbl = (unsigned)(uintptr_t)(void*)&sbl[0][0];

    // ---- stage the 64-row x-tile (hi and lo) into LDS ----
    {
        const uint4* gh = (const uint4*)(xh + (size_t)rowBase * DIM);
        const uint4* gl = (const uint4*)(xl + (size_t)rowBase * DIM);
        uint4* sh = (uint4*)sxh;
        uint4* sl = (uint4*)sxl;
        const int total = ROWS_PER_BLOCK * DIM / 8;   // #uint4 = 4096
        #pragma unroll 1
        for (int i = tid; i < total; i += ARGMIN_THREADS) {
            sh[i] = gh[i];
            sl[i] = gl[i];
        }
    }

    // ---- issue async DMA for B chunk 0 ----
    // chunk c: codes [(c>>3)*64, +64), dims [(c&7)*64, +64); 16 KB hi+lo
    {
        const int kb = 0, db = 0;
        #pragma unroll
        for (int j2 = 0; j2 < 4; ++j2) {
            const int j   = tid + j2 * ARGMIN_THREADS;   // uint4 id in [0,512)
            const int row = j >> 3;                      // code row 0..63
            const int col = (j & 7) * 8;                 // dim start
            const size_t goff = (size_t)(kb + row) * DIM + db + col;
            const unsigned loff = (unsigned)(j * 16);
            async_ld_b128(ldsbh + loff, eh + goff);
            async_ld_b128(ldsbl + loff, el + goff);
        }
    }

    const int waveRow = wave * 16;           // this wave's 16-row stripe
    const int lrow = lane & 15;              // A: row in stripe / B: code column
    const int oct  = (lane >> 4) * 8;        // K-octet select per 16-lane half

    float minv[8];
    int   minidx[8];
    #pragma unroll
    for (int j = 0; j < 8; ++j) { minv[j] = 3.4e38f; minidx[j] = 0; }

    #pragma unroll 1
    for (int kbi = 0; kbi < N_KB; ++kbi) {
        v8f acc[4];
        #pragma unroll
        for (int t = 0; t < 4; ++t) acc[t] = (v8f){};

        #pragma unroll 1
        for (int dbi = 0; dbi < N_DB; ++dbi) {
            const int c = kbi * N_DB + dbi;

            // Drain this wave's outstanding DMA, then make chunk c visible to
            // all waves (barrier also proves everyone is done reading the
            // buffer chunk c+1 will overwrite).
            wait_asynccnt0();
            __syncthreads();

            // Prefetch chunk c+1 into the other buffer (overlaps compute).
            if (c + 1 < N_CHUNKS) {
                const int nkb = ((c + 1) >> 3) * B_CODES;
                const int ndb = ((c + 1) & 7) * B_DIMS;
                const unsigned bufoff = (unsigned)(((c + 1) & 1) * B_CODES * B_DIMS * 2);
                #pragma unroll
                for (int j2 = 0; j2 < 4; ++j2) {
                    const int j   = tid + j2 * ARGMIN_THREADS;
                    const int row = j >> 3;
                    const int col = (j & 7) * 8;
                    const size_t goff = (size_t)(nkb + row) * DIM + ndb + col;
                    const unsigned loff = bufoff + (unsigned)(j * 16);
                    async_ld_b128(ldsbh + loff, eh + goff);
                    async_ld_b128(ldsbl + loff, el + goff);
                }
            }

            // ---- consume chunk c: 2 x (4 tiles x 3 WMMAs) ----
            const int cur = c & 1;
            const int db  = dbi * B_DIMS;
            #pragma unroll
            for (int ddi = 0; ddi < 2; ++ddi) {
                const int dd = ddi * 32;
                V16 ah, al;
                const size_t abase = (size_t)(waveRow + lrow) * DIM + db + dd + oct;
                ah.q[0] = *(const uint4*)&sxh[abase];
                ah.q[1] = *(const uint4*)&sxh[abase + 16];
                al.q[0] = *(const uint4*)&sxl[abase];
                al.q[1] = *(const uint4*)&sxl[abase + 16];

                #pragma unroll
                for (int t = 0; t < 4; ++t) {
                    const int bbase = (t * 16 + lrow) * B_DIMS + dd + oct;
                    V16 bh, bl;
                    bh.q[0] = *(const uint4*)&sbh[cur][bbase];
                    bh.q[1] = *(const uint4*)&sbh[cur][bbase + 16];
                    bl.q[0] = *(const uint4*)&sbl[cur][bbase];
                    bl.q[1] = *(const uint4*)&sbl[cur][bbase + 16];

                    acc[t] = __builtin_amdgcn_wmma_f32_16x16x32_bf16(
                        false, ah.v, false, bh.v, (short)0, acc[t], false, false);
                    acc[t] = __builtin_amdgcn_wmma_f32_16x16x32_bf16(
                        false, ah.v, false, bl.v, (short)0, acc[t], false, false);
                    acc[t] = __builtin_amdgcn_wmma_f32_16x16x32_bf16(
                        false, al.v, false, bh.v, (short)0, acc[t], false, false);
                }
            }
        }

        // score = |e|^2 - 2*x.e ; update running argmin.
        // C layout: VGPR j, lane l -> row = j + 8*(l>=16), col = l&15
        #pragma unroll
        for (int t = 0; t < 4; ++t) {
            const int code = kbi * B_CODES + t * 16 + lrow;
            const float e2 = en2[code];
            #pragma unroll
            for (int j = 0; j < 8; ++j) {
                float v = e2 - 2.0f * acc[t][j];
                bool better = (v < minv[j]) || (v == minv[j] && code < minidx[j]);
                if (better) { minv[j] = v; minidx[j] = code; }
            }
        }
    }

    // Cross-lane min over the 16 lanes that share a row (stay inside half-wave).
    #pragma unroll
    for (int j = 0; j < 8; ++j) {
        float v = minv[j];
        int   ix = minidx[j];
        #pragma unroll
        for (int m = 8; m >= 1; m >>= 1) {
            float ov = __shfl_xor(v, m, 32);
            int   oi = __shfl_xor(ix, m, 32);
            if (ov < v || (ov == v && oi < ix)) { v = ov; ix = oi; }
        }
        if (lane == 0)  idx_out[rowBase + waveRow + j]     = ix;
        if (lane == 16) idx_out[rowBase + waveRow + 8 + j] = ix;
    }
}

// ---------------- Kernel 3: gather quantized, loss partials, histogram ----------------
__global__ void vq_gather(const float* __restrict__ x, const float* __restrict__ cb,
                          const int* __restrict__ idx,
                          float* __restrict__ out_q, float* __restrict__ out_idx_f,
                          int* __restrict__ counts, float* __restrict__ loss_acc) {
    const int n = blockIdx.x;
    const int c = idx[n];
    const float* q  = cb + (size_t)c * DIM;
    const float* xi = x  + (size_t)n * DIM;
    float* o = out_q + (size_t)n * DIM;

    float s = 0.f;
    for (int d = threadIdx.x; d < DIM; d += blockDim.x) {
        float qv = q[d];
        float df = qv - xi[d];
        o[d] = qv;                  // quantized_st value == quantized
        s += df * df;
    }
    __shared__ float red[8];
    #pragma unroll
    for (int m = 16; m >= 1; m >>= 1) s += __shfl_xor(s, m, 32);
    const int lane = threadIdx.x & 31, w = threadIdx.x >> 5;
    if (lane == 0) red[w] = s;
    __syncthreads();
    if (threadIdx.x == 0) {
        float t = 0.f;
        const int nw = blockDim.x >> 5;
        for (int i = 0; i < nw; ++i) t += red[i];
        atomicAdd(loss_acc, t);
        atomicAdd(&counts[c], 1);
        out_idx_f[n] = (float)c;
    }
}

// ---------------- Kernel 4: finalize scalars ----------------
__global__ void vq_finalize(const int* __restrict__ counts,
                            const float* __restrict__ loss_acc,
                            float* __restrict__ out_scalars) {
    __shared__ float red[8];
    float s = 0.f;
    for (int c = threadIdx.x; c < K_CODES; c += blockDim.x) {
        float p = (float)counts[c] / (float)N_ROWS;
        s += p * __logf(p + 1e-10f);
    }
    #pragma unroll
    for (int m = 16; m >= 1; m >>= 1) s += __shfl_xor(s, m, 32);
    const int lane = threadIdx.x & 31, w = threadIdx.x >> 5;
    if (lane == 0) red[w] = s;
    __syncthreads();
    if (threadIdx.x == 0) {
        float t = 0.f;
        const int nw = blockDim.x >> 5;
        for (int i = 0; i < nw; ++i) t += red[i];
        // loss = q_latent + 0.25 * e_latent, both numerically identical
        out_scalars[0] = 1.25f * loss_acc[0] / (float)((size_t)N_ROWS * DIM);
        out_scalars[1] = __expf(-t);   // perplexity
    }
}

extern "C" void kernel_launch(void* const* d_in, const int* in_sizes, int n_in,
                              void* d_out, int out_size, void* d_ws, size_t ws_size,
                              hipStream_t stream) {
    const float* x  = (const float*)d_in[0];   // [32768, 512] fp32
    const float* cb = (const float*)d_in[1];   // [4096, 512]  fp32
    float* out = (float*)d_out;
    char*  ws  = (char*)d_ws;

    unsigned short* xh = (unsigned short*)(ws + OFF_XH);
    unsigned short* xl = (unsigned short*)(ws + OFF_XL);
    unsigned short* eh = (unsigned short*)(ws + OFF_EH);
    unsigned short* el = (unsigned short*)(ws + OFF_EL);
    float* en2      = (float*)(ws + OFF_EN2);
    int*   idx_i    = (int*)(ws + OFF_IDX);
    int*   counts   = (int*)(ws + OFF_CNT);
    float* loss_acc = (float*)(ws + OFF_ACC);

    // Output layout: [quantized_st 32768*512][loss][perplexity][indices 32768]
    float* out_q       = out;
    float* out_scalars = out + (size_t)N_ROWS * DIM;
    float* out_idx_f   = out + (size_t)N_ROWS * DIM + 2;

    // 1) precision splits + norms + zeroed accumulators
    vq_split_bf16<<<(N_ROWS * DIM + 255) / 256, 256, 0, stream>>>(x, xh, xl, N_ROWS * DIM);
    vq_split_bf16<<<(K_CODES * DIM + 255) / 256, 256, 0, stream>>>(cb, eh, el, K_CODES * DIM);
    vq_enorm<<<K_CODES / 256, 256, 0, stream>>>(cb, en2);
    vq_zero<<<(K_CODES + 255) / 256, 256, 0, stream>>>(counts, loss_acc);

    // 2) WMMA distance GEMM + argmin (async-DMA B staging)
    vq_argmin_wmma<<<N_ROWS / ROWS_PER_BLOCK, ARGMIN_THREADS, 0, stream>>>(
        xh, xl, eh, el, en2, idx_i);

    // 3) gather + loss + histogram
    vq_gather<<<N_ROWS, 256, 0, stream>>>(x, cb, idx_i, out_q, out_idx_f, counts, loss_acc);

    // 4) scalars
    vq_finalize<<<1, 256, 0, stream>>>(counts, loss_acc, out_scalars);
}
